// GAE_13039520711025
// MI455X (gfx1250) — matrix operand order repaired and checked by reading, not compiled, not run on hardware
//
#include <hip/hip_runtime.h>
#include <hip/hip_bf16.h>

typedef __attribute__((ext_vector_type(16))) __bf16 v16bf;
typedef __attribute__((ext_vector_type(8)))  __bf16 bf16x8;
typedef __attribute__((ext_vector_type(8)))  float  v8f;

static __device__ __forceinline__ __bf16 tobf(float f) {
    return (__bf16)f;   // RNE convert, lowers to v_cvt on gfx1250
}

// Epilogue modes:
// 0: store bf16 to out0
// 1: +bias, relu, store bf16 to out0
// 2: +bias, store fp32 to out0 AND bf16 to out1
// 3: sigmoid, store fp32 to out0
template<int TNT, bool A_BF16, bool B_BF16, int EPI>
__global__ __launch_bounds__(256) void gemm_wmma_bf16(
    const void* __restrict__ Ap, const void* __restrict__ Bp,
    const float* __restrict__ bias,
    void* __restrict__ out0, void* __restrict__ out1,
    int M, int N, int K, int lda, int ldb, int ldo)
{
    const int lane = threadIdx.x & 31;
    const int wid  = (blockIdx.x * blockDim.x + threadIdx.x) >> 5;
    const int ngroups = N / (16 * TNT);
    const int total   = (M / 16) * ngroups;
    if (wid >= total) return;                 // whole-wave uniform exit

    const int tm = wid / ngroups;
    const int ng = wid % ngroups;
    const int m0 = tm * 16;
    const int n0 = ng * (16 * TNT);
    const int half = lane >> 4;               // 0 or 1
    const int l15  = lane & 15;
    const int m    = m0 + l15;                // A-fragment row for this lane

    v8f acc[TNT];
#pragma unroll
    for (int t = 0; t < TNT; ++t)
        acc[t] = v8f{0.f,0.f,0.f,0.f,0.f,0.f,0.f,0.f};

    for (int k0 = 0; k0 < K; k0 += 32) {
        // ---- A fragment: lane holds row m, K = k0 + 8*half + [0..7] and +16 ----
        v16bf a;
        if constexpr (A_BF16) {
            const __bf16* A = (const __bf16*)Ap;
            const __bf16* p = A + (size_t)m * lda + k0 + half * 8;
            bf16x8 lo = *(const bf16x8*)(p);
            bf16x8 hi = *(const bf16x8*)(p + 16);
#pragma unroll
            for (int e = 0; e < 8; ++e) { a[e] = lo[e]; a[8 + e] = hi[e]; }
        } else {
            const float* A = (const float*)Ap;
            const float* p = A + (size_t)m * lda + k0 + half * 8;
            float4 x0 = *(const float4*)(p);
            float4 x1 = *(const float4*)(p + 4);
            float4 x2 = *(const float4*)(p + 16);
            float4 x3 = *(const float4*)(p + 20);
            a[0]=tobf(x0.x); a[1]=tobf(x0.y); a[2]=tobf(x0.z); a[3]=tobf(x0.w);
            a[4]=tobf(x1.x); a[5]=tobf(x1.y); a[6]=tobf(x1.z); a[7]=tobf(x1.w);
            a[8]=tobf(x2.x); a[9]=tobf(x2.y); a[10]=tobf(x2.z); a[11]=tobf(x2.w);
            a[12]=tobf(x3.x); a[13]=tobf(x3.y); a[14]=tobf(x3.z); a[15]=tobf(x3.w);
        }

        const int kb = k0 + half * 8;         // B-fragment base K for this lane (K = kb + [0..7] and +16)
#pragma unroll
        for (int t = 0; t < TNT; ++t) {
            const int nb = n0 + t * 16;
            v16bf b;
            if constexpr (B_BF16) {
                const __bf16* B = (const __bf16*)Bp;
                const __bf16* p = B + (size_t)kb * ldb + nb + l15;
#pragma unroll
                for (int e = 0; e < 8; ++e) { b[e] = p[(size_t)e * ldb]; b[8 + e] = p[(size_t)(16 + e) * ldb]; }
            } else {
                const float* B = (const float*)Bp;
                const float* p = B + (size_t)kb * ldb + nb + l15;
#pragma unroll
                for (int e = 0; e < 8; ++e) { b[e] = tobf(p[(size_t)e * ldb]); b[8 + e] = tobf(p[(size_t)(16 + e) * ldb]); }
            }
            acc[t] = __builtin_amdgcn_wmma_f32_16x16x32_bf16(
                         false, a, false, b, (short)0, acc[t], false, false);
        }
    }

    // ---- Epilogue: D VGPR v -> row m0 + v + 8*half, col n0 + t*16 + l15 ----
#pragma unroll
    for (int t = 0; t < TNT; ++t) {
        const int c = n0 + t * 16 + l15;
        float bv = 0.f;
        if constexpr (EPI == 1 || EPI == 2) bv = bias[c];
#pragma unroll
        for (int v = 0; v < 8; ++v) {
            const int r = m0 + v + 8 * half;
            const size_t idx = (size_t)r * ldo + c;
            float val = acc[t][v];
            if constexpr (EPI == 1) {
                val += bv; val = val > 0.f ? val : 0.f;
                ((__bf16*)out0)[idx] = tobf(val);
            } else if constexpr (EPI == 2) {
                val += bv;
                ((float*)out0)[idx] = val;
                ((__bf16*)out1)[idx] = tobf(val);
            } else if constexpr (EPI == 3) {
                val = 1.f / (1.f + __expf(-val));
                ((float*)out0)[idx] = val;
            } else {
                ((__bf16*)out0)[idx] = tobf(val);
            }
        }
    }
}

__global__ __launch_bounds__(256) void transpose_bf16(
    const __bf16* __restrict__ in, __bf16* __restrict__ out, int rows, int cols)
{
    int idx = blockIdx.x * blockDim.x + threadIdx.x;
    if (idx >= rows * cols) return;
    int r = idx / cols;
    int c = idx - r * cols;
    out[(size_t)c * rows + r] = in[idx];
}

extern "C" void kernel_launch(void* const* d_in, const int* in_sizes, int n_in,
                              void* d_out, int out_size, void* d_ws, size_t ws_size,
                              hipStream_t stream)
{
    constexpr int N = 8192, IN = 512, HID = 256, OUT = 128;

    const float* adj     = (const float*)d_in[0];
    const float* feature = (const float*)d_in[1];
    const float* W1      = (const float*)d_in[2];
    const float* b1      = (const float*)d_in[3];
    const float* W2      = (const float*)d_in[4];
    const float* b2      = (const float*)d_in[5];

    float* out_noise = (float*)d_out;                       // N*N sigmoid logits
    float* out_h     = out_noise + (size_t)N * N;           // N*OUT fp32 h2

    // bf16 workspace (all L2-resident): 14 MB total
    char* ws = (char*)d_ws;
    __bf16* xw1 = (__bf16*)(ws);                            // [N][HID]  4 MB
    __bf16* h1b = (__bf16*)(ws + (size_t)4  * 1024 * 1024); // [N][HID]  4 MB
    __bf16* hw2 = (__bf16*)(ws + (size_t)8  * 1024 * 1024); // [N][OUT]  2 MB
    __bf16* h2b = (__bf16*)(ws + (size_t)10 * 1024 * 1024); // [N][OUT]  2 MB
    __bf16* h2t = (__bf16*)(ws + (size_t)12 * 1024 * 1024); // [OUT][N]  2 MB

    auto blocks_for = [](long long waves) { return (int)((waves * 32 + 255) / 256); };

    // 1) XW1 = feature @ W1                    [8192,512]x[512,256] -> bf16
    {
        long long waves = (long long)(N / 16) * (HID / 64);
        gemm_wmma_bf16<4, false, false, 0><<<blocks_for(waves), 256, 0, stream>>>(
            feature, W1, nullptr, xw1, nullptr, N, HID, IN, IN, HID, HID);
    }
    // 2) h1 = relu(adj @ XW1 + b1)             [8192,8192]x[8192,256] -> bf16
    {
        long long waves = (long long)(N / 16) * (HID / 64);
        gemm_wmma_bf16<4, false, true, 1><<<blocks_for(waves), 256, 0, stream>>>(
            adj, xw1, b1, h1b, nullptr, N, HID, N, N, HID, HID);
    }
    // 3) HW2 = h1 @ W2                         [8192,256]x[256,128] -> bf16
    {
        long long waves = (long long)(N / 16) * (OUT / 64);
        gemm_wmma_bf16<4, true, false, 0><<<blocks_for(waves), 256, 0, stream>>>(
            h1b, W2, nullptr, hw2, nullptr, N, OUT, HID, HID, OUT, OUT);
    }
    // 4) h2 = adj @ HW2 + b2  -> fp32 (d_out tail) + bf16 (ws)
    {
        long long waves = (long long)(N / 16) * (OUT / 64);
        gemm_wmma_bf16<4, false, true, 2><<<blocks_for(waves), 256, 0, stream>>>(
            adj, hw2, b2, out_h, h2b, N, OUT, N, N, OUT, OUT);
    }
    // 5) h2t = h2^T (bf16) so decoder B-loads are contiguous
    {
        int total = N * OUT;
        transpose_bf16<<<(total + 255) / 256, 256, 0, stream>>>(h2b, h2t, N, OUT);
    }
    // 6) out_noise = sigmoid(h2 @ h2^T)        [8192,128]x[128,8192] -> fp32
    {
        long long waves = (long long)(N / 16) * (N / 64);
        gemm_wmma_bf16<4, true, true, 3><<<blocks_for(waves), 256, 0, stream>>>(
            h2b, h2t, nullptr, out_noise, nullptr, N, N, OUT, OUT, N, N);
    }
}
